// ExtraMSABlock_40123584479678
// MI455X (gfx1250) — compile-verified, matrix operand order checked by reading, not run on hardware
//
#include <hip/hip_runtime.h>
#include <hip/hip_bf16.h>

// ===========================================================================
// ExtraMSA block for MI455X (gfx1250). All GEMM math via v_wmma_f32_16x16x32_f16
// (wave32). Fast path: 32x32 tile/wave, software-pipelined k-loop, vectorized
// b128 fragment loads (A k-contiguous activations, weights pre-transposed to
// [N,K] so B is k-contiguous too). Safe path: bounds-checked 16x16 tile for
// ragged shapes. Attention logits processed in ~33MB chunks (L2-resident).
// ===========================================================================

typedef __attribute__((ext_vector_type(16))) _Float16 v16h;
typedef __attribute__((ext_vector_type(8)))  _Float16 v8h;
typedef __attribute__((ext_vector_type(8)))  float    v8f;

// Fragment loader per CDNA5 ISA 7.12.2 16-bit layout:
// element pair (2j,2j+1) holds K = k0 + (j&4?16:0) + hi*8 + 2*(j&3) (+1).
// For ks==1 this is two contiguous 16-byte runs -> two b128 loads.
__device__ __forceinline__ v16h load_frag(const _Float16* __restrict__ base,
                                          long fixs, long ks, int idx, int k0, int hi)
{
    if (ks == 1) {
        const _Float16* p = base + (long)idx * fixs + k0 + hi * 8;
        const v8h lo = *(const v8h*)p;
        const v8h hh = *(const v8h*)(p + 16);
        return __builtin_shufflevector(lo, hh, 0,1,2,3,4,5,6,7,8,9,10,11,12,13,14,15);
    }
    v16h f;
#pragma unroll
    for (int j = 0; j < 8; ++j) {
        const int kk = k0 + ((j & 4) ? 16 : 0) + hi * 8 + ((j & 3) << 1);
        f[2*j]   = base[(long)idx * fixs + (long)kk * ks];
        f[2*j+1] = base[(long)idx * fixs + (long)(kk + 1) * ks];
    }
    return f;
}

__device__ __forceinline__ void store_tile(v8f acc, int mb, int nb, int lm, int hi,
                                           float* __restrict__ C, _Float16* __restrict__ C16,
                                           long co, long rsC, long csC,
                                           const float* __restrict__ bias,
                                           const float* __restrict__ R, float scale, int act)
{
    const int n = nb + lm;
#pragma unroll
    for (int r = 0; r < 8; ++r) {
        const int m = mb + r + 8 * hi;           // C/D layout: VGPR r, half-wave hi
        float v = acc[r] * scale;
        if (bias) v += bias[n];
        if (act == 1)      v = v > 0.f ? v : 0.f;
        else if (act == 2) v = 1.f / (1.f + __expf(-v));
        const long off = co + (long)m * rsC + (long)n * csC;
        if (R)   v += R[off];
        if (C)   C[off]   = v;
        if (C16) C16[off] = (_Float16)v;
    }
}

// ------------------- fast path: 32x32 tile per wave ------------------------
// requires M%32==0, N%32==0, K%32==0. Block (32,4): 4 waves tile 64x64.
// k-loop software-pipelined: next k-step's fragments are issued before the
// current WMMAs so loads overlap matrix ops instead of draining LOADcnt.
__global__ void __launch_bounds__(128)
wmma_gemm32_kernel(const _Float16* __restrict__ A, long rsA, long csA, long a1, long a2,
                   const _Float16* __restrict__ B, long rsB, long csB, long b1, long b2,
                   float* __restrict__ C, _Float16* __restrict__ C16,
                   long rsC, long csC, long c1, long c2,
                   const float* __restrict__ bias, const float* __restrict__ R,
                   int M, int N, int K, int nb2, float scale, int act)
{
    const int lane = threadIdx.x, ty = threadIdx.y;
    const int m0 = blockIdx.x * 64 + (ty >> 1) * 32;
    const int n0 = blockIdx.y * 64 + (ty & 1) * 32;
    if (m0 >= M || n0 >= N) return;              // wave-uniform
    const int bq = blockIdx.z / nb2, br = blockIdx.z % nb2;
    const _Float16* Ab = A + (long)bq * a1 + (long)br * a2;
    const _Float16* Bb = B + (long)bq * b1 + (long)br * b2;
    const long co = (long)bq * c1 + (long)br * c2;
    const int hi = lane >> 4, lm = lane & 15;

    v8f acc00 = {}, acc01 = {}, acc10 = {}, acc11 = {};
    v16h a0 = load_frag(Ab, rsA, csA, m0 + lm,      0, hi);
    v16h a1f= load_frag(Ab, rsA, csA, m0 + 16 + lm, 0, hi);
    v16h b0 = load_frag(Bb, csB, rsB, n0 + lm,      0, hi);
    v16h b1f= load_frag(Bb, csB, rsB, n0 + 16 + lm, 0, hi);
    for (int k0 = 0; k0 < K; k0 += 32) {
        v16h na0 = a0, na1 = a1f, nb0 = b0, nb1 = b1f;
        const bool more = (k0 + 32 < K);
        if (more) {                              // issue next k-step's loads first
            na0 = load_frag(Ab, rsA, csA, m0 + lm,      k0 + 32, hi);
            na1 = load_frag(Ab, rsA, csA, m0 + 16 + lm, k0 + 32, hi);
            nb0 = load_frag(Bb, csB, rsB, n0 + lm,      k0 + 32, hi);
            nb1 = load_frag(Bb, csB, rsB, n0 + 16 + lm, k0 + 32, hi);
            if (csA == 1)                        // global_prefetch_b8 k+64 A tile
                __builtin_prefetch(Ab + (long)(m0 + lm) * rsA + k0 + 64, 0, 0);
        }
        acc00 = __builtin_amdgcn_wmma_f32_16x16x32_f16(false, a0,  false, b0,  (short)0, acc00, false, false);
        acc01 = __builtin_amdgcn_wmma_f32_16x16x32_f16(false, a0,  false, b1f, (short)0, acc01, false, false);
        acc10 = __builtin_amdgcn_wmma_f32_16x16x32_f16(false, a1f, false, b0,  (short)0, acc10, false, false);
        acc11 = __builtin_amdgcn_wmma_f32_16x16x32_f16(false, a1f, false, b1f, (short)0, acc11, false, false);
        a0 = na0; a1f = na1; b0 = nb0; b1f = nb1;
    }
    store_tile(acc00, m0,      n0,      lm, hi, C, C16, co, rsC, csC, bias, R, scale, act);
    store_tile(acc01, m0,      n0 + 16, lm, hi, C, C16, co, rsC, csC, bias, R, scale, act);
    store_tile(acc10, m0 + 16, n0,      lm, hi, C, C16, co, rsC, csC, bias, R, scale, act);
    store_tile(acc11, m0 + 16, n0 + 16, lm, hi, C, C16, co, rsC, csC, bias, R, scale, act);
}

// ------------- safe path: bounds-checked 16x16 tile per wave ----------------
__global__ void __launch_bounds__(128)
wmma_gemm16_kernel(const _Float16* __restrict__ A, long rsA, long csA, long a1, long a2,
                   const _Float16* __restrict__ B, long rsB, long csB, long b1, long b2,
                   float* __restrict__ C, _Float16* __restrict__ C16,
                   long rsC, long csC, long c1, long c2,
                   const float* __restrict__ bias, const float* __restrict__ R,
                   int M, int N, int K, int nb2, float scale, int act)
{
    const int lane = threadIdx.x;
    const int m0 = blockIdx.x * 16;
    const int n0 = (blockIdx.y * 4 + threadIdx.y) * 16;
    if (n0 >= N) return;
    const int bq = blockIdx.z / nb2, br = blockIdx.z % nb2;
    const _Float16* Ab = A + (long)bq * a1 + (long)br * a2;
    const _Float16* Bb = B + (long)bq * b1 + (long)br * b2;
    const long co = (long)bq * c1 + (long)br * c2;
    const int hi = lane >> 4, lm = lane & 15;
    const int am = m0 + lm, bn = n0 + lm;

    v8f acc = {};
    for (int k0 = 0; k0 < K; k0 += 32) {
        v16h af, bf;
#pragma unroll
        for (int i = 0; i < 8; ++i) {
            const int kk = k0 + ((i & 4) ? 16 : 0) + hi * 8 + ((i & 3) << 1);
            _Float16 x0 = (_Float16)0.f, x1 = (_Float16)0.f;
            if (am < M) {
                if (kk < K)     x0 = Ab[(long)am * rsA + (long)kk * csA];
                if (kk + 1 < K) x1 = Ab[(long)am * rsA + (long)(kk + 1) * csA];
            }
            af[2*i] = x0; af[2*i+1] = x1;
            _Float16 y0 = (_Float16)0.f, y1 = (_Float16)0.f;
            if (bn < N) {
                if (kk < K)     y0 = Bb[(long)kk * rsB + (long)bn * csB];
                if (kk + 1 < K) y1 = Bb[(long)(kk + 1) * rsB + (long)bn * csB];
            }
            bf[2*i] = y0; bf[2*i+1] = y1;
        }
        acc = __builtin_amdgcn_wmma_f32_16x16x32_f16(false, af, false, bf, (short)0, acc, false, false);
    }
    if (bn < N) {
#pragma unroll
        for (int r = 0; r < 8; ++r) {
            const int m = m0 + r + 8 * hi;
            if (m < M) {
                float v = acc[r] * scale;
                if (bias) v += bias[bn];
                if (act == 1)      v = v > 0.f ? v : 0.f;
                else if (act == 2) v = 1.f / (1.f + __expf(-v));
                const long off = co + (long)m * rsC + (long)bn * csC;
                if (R)   v += R[off];
                if (C)   C[off]   = v;
                if (C16) C16[off] = (_Float16)v;
            }
        }
    }
}

// ------------------------- LayerNorm (D <= 128) ----------------------------
__global__ void __launch_bounds__(128)
ln_kernel(const float* __restrict__ x, const float* __restrict__ g,
          const float* __restrict__ b, _Float16* __restrict__ y, int D)
{
    const long row = blockIdx.x;
    const float* xr = x + row * D;
    __shared__ float red[128];
    const int t = threadIdx.x;
    const float v = (t < D) ? xr[t] : 0.f;
    red[t] = v; __syncthreads();
    for (int s = 64; s > 0; s >>= 1) { if (t < s) red[t] += red[t + s]; __syncthreads(); }
    const float mu = red[0] / D;
    __syncthreads();
    const float d = (t < D) ? (v - mu) : 0.f;
    red[t] = d * d; __syncthreads();
    for (int s = 64; s > 0; s >>= 1) { if (t < s) red[t] += red[t + s]; __syncthreads(); }
    const float rstd = rsqrtf(red[0] / D + 1e-5f);
    if (t < D) y[row * D + t] = (_Float16)((v - mu) * rstd * g[t] + b[t]);
}

// --------------- Strided softmax (L <= 256), writes f16 ---------------------
__global__ void __launch_bounds__(256)
softmax_kernel(const float* __restrict__ x, _Float16* __restrict__ y,
               int L, long rstride, long estride)
{
    const long row = blockIdx.x;
    const float* xb = x + row * rstride;
    _Float16*    yb = y + row * rstride;
    __shared__ float red[256];
    const int t = threadIdx.x;
    const float v = (t < L) ? xb[(long)t * estride] : -3.0e38f;
    red[t] = v; __syncthreads();
    for (int s = 128; s > 0; s >>= 1) { if (t < s) red[t] = fmaxf(red[t], red[t + s]); __syncthreads(); }
    const float mx = red[0];
    __syncthreads();
    const float e = (t < L) ? __expf(v - mx) : 0.f;
    red[t] = e; __syncthreads();
    for (int s = 128; s > 0; s >>= 1) { if (t < s) red[t] += red[t + s]; __syncthreads(); }
    const float inv = 1.f / red[0];
    if (t < L) yb[(long)t * estride] = (_Float16)(e * inv);
}

// ----------------------------- elementwise ---------------------------------
__global__ void cvt16_kernel(const float* __restrict__ a, _Float16* __restrict__ o, long n) {
    for (long i = blockIdx.x * (long)blockDim.x + threadIdx.x; i < n; i += (long)gridDim.x * blockDim.x)
        o[i] = (_Float16)a[i];
}
// weights [K,N] (row-major) -> transposed f16 [N,K] so GEMM-B is k-contiguous
__global__ void cvtT16_kernel(const float* __restrict__ a, _Float16* __restrict__ o, long K, long N) {
    const long n = K * N;
    for (long i = blockIdx.x * (long)blockDim.x + threadIdx.x; i < n; i += (long)gridDim.x * blockDim.x) {
        const long k = i / N, c = i % N;
        o[c * K + k] = (_Float16)a[i];
    }
}
__global__ void copy_kernel(const float* __restrict__ a, float* __restrict__ o, long n) {
    for (long i = blockIdx.x * (long)blockDim.x + threadIdx.x; i < n; i += (long)gridDim.x * blockDim.x)
        o[i] = a[i];
}
__global__ void mul_ff16_kernel(const float* __restrict__ a, const float* __restrict__ b,
                                _Float16* __restrict__ o, long n) {
    for (long i = blockIdx.x * (long)blockDim.x + threadIdx.x; i < n; i += (long)gridDim.x * blockDim.x)
        o[i] = (_Float16)(a[i] * b[i]);
}
__global__ void bcast_mul16_kernel(const float* __restrict__ a, const float* __restrict__ b,
                                   _Float16* __restrict__ o, long n, long mod) {
    for (long i = blockIdx.x * (long)blockDim.x + threadIdx.x; i < n; i += (long)gridDim.x * blockDim.x)
        o[i] = (_Float16)(a[i] * b[i % mod]);
}
__global__ void mul_h16f_kernel(const _Float16* __restrict__ g, const float* __restrict__ o,
                                _Float16* __restrict__ out, long n) {
    for (long i = blockIdx.x * (long)blockDim.x + threadIdx.x; i < n; i += (long)gridDim.x * blockDim.x)
        out[i] = (_Float16)((float)g[i] * o[i]);
}
__global__ void mul_gate_jchunk_kernel(const _Float16* __restrict__ g, const float* __restrict__ o,
                                       _Float16* __restrict__ out, int j0, int CH, long n) {
    for (long i = blockIdx.x * (long)blockDim.x + threadIdx.x; i < n; i += (long)gridDim.x * blockDim.x) {
        const long r = i / 512, x = i % 512;
        const long ii = r / CH, jl = r % CH;
        out[i] = (_Float16)((float)g[(ii * 256 + j0 + jl) * 512 + x] * o[i]);
    }
}
__global__ void gated_add_kernel(float* __restrict__ dst, const float* __restrict__ g,
                                 const float* __restrict__ t, long n) {
    for (long i = blockIdx.x * (long)blockDim.x + threadIdx.x; i < n; i += (long)gridDim.x * blockDim.x)
        dst[i] += g[i] * t[i];
}
__global__ void scatter_add_jchunk_kernel(float* __restrict__ dst, const float* __restrict__ t,
                                          int j0, int CH, long n) {
    for (long i = blockIdx.x * (long)blockDim.x + threadIdx.x; i < n; i += (long)gridDim.x * blockDim.x) {
        const long r = i / 128, col = i % 128;
        const long ii = r / CH, jl = r % CH;
        dst[(ii * 256 + j0 + jl) * 128 + col] += t[i];
    }
}
__global__ void addbias_rowattn_kernel(float* __restrict__ lg, const float* __restrict__ brow, long n) {
    for (long idx = blockIdx.x * (long)blockDim.x + threadIdx.x; idx < n; idx += (long)gridDim.x * blockDim.x) {
        const long j = idx % 256, i = (idx / 256) % 256, h = (idx / 65536) % 8;
        lg[idx] += brow[(i * 256 + j) * 8 + h];
    }
}
__global__ void addbias_tri_kernel(float* __restrict__ lg, const float* __restrict__ bt,
                                   int endmode, long n) {
    for (long idx = blockIdx.x * (long)blockDim.x + threadIdx.x; idx < n; idx += (long)gridDim.x * blockDim.x) {
        const long k = idx % 256, i2 = (idx / 256) % 256, h = (idx / 65536) % 4;
        lg[idx] += endmode ? bt[(k * 256 + i2) * 4 + h] : bt[(i2 * 256 + k) * 4 + h];
    }
}
__global__ void mean_s_kernel(const float* __restrict__ q, _Float16* __restrict__ o,
                              int S, int I, int ld, int ncols) {
    const long n = (long)I * ncols;
    for (long idx = blockIdx.x * (long)blockDim.x + threadIdx.x; idx < n; idx += (long)gridDim.x * blockDim.x) {
        const long i = idx / ncols, x = idx % ncols;
        float s = 0.f;
        for (int t = 0; t < S; ++t) s += q[((long)t * I + i) * ld + x];
        o[idx] = (_Float16)(s / S);
    }
}

// ---------------------------------------------------------------------------
static inline unsigned gcount(long n) { long g = (n + 255) / 256; return (unsigned)(g > 0 ? g : 1); }

static inline void gemm(hipStream_t st,
                        const _Float16* A, long rsA, long csA, long a1, long a2,
                        const _Float16* B, long rsB, long csB, long b1, long b2,
                        float* C, _Float16* C16, long rsC, long csC, long c1, long c2,
                        const float* bias, const float* R,
                        long M, long N, long K, int nbatch, int nb2, float scale, int act)
{
    const int nb = nb2 < 1 ? 1 : nb2;
    if ((M % 32) == 0 && (N % 32) == 0 && (K % 32) == 0) {
        dim3 g((unsigned)((M + 63) / 64), (unsigned)((N + 63) / 64), (unsigned)nbatch);
        wmma_gemm32_kernel<<<g, dim3(32, 4, 1), 0, st>>>(A, rsA, csA, a1, a2, B, rsB, csB, b1, b2,
            C, C16, rsC, csC, c1, c2, bias, R, (int)M, (int)N, (int)K, nb, scale, act);
    } else {
        dim3 g((unsigned)((M + 15) / 16), (unsigned)((N + 63) / 64), (unsigned)nbatch);
        wmma_gemm16_kernel<<<g, dim3(32, 4, 1), 0, st>>>(A, rsA, csA, a1, a2, B, rsB, csB, b1, b2,
            C, C16, rsC, csC, c1, c2, bias, R, (int)M, (int)N, (int)K, nb, scale, act);
    }
}

// Parameter leaf order = JAX pytree (sorted dict keys at every level).
enum ParamIdx {
    IN_MSA = 0, IN_PAIR = 1,
    GLOB_GATE_B = 2, GLOB_GATE_W, GLOB_LN_B, GLOB_LN_G, GLOB_OUT_B, GLOB_OUT_W, GLOB_QKV_W,
    MTR_LN_B, MTR_LN_G, MTR_W1_B, MTR_W1_W, MTR_W2_B, MTR_W2_W,
    OPM_A_B, OPM_A_W, OPM_B_B, OPM_B_W, OPM_LN_B, OPM_LN_G, OPM_OUT_B, OPM_OUT_W,
    PTR_LN_B, PTR_LN_G, PTR_W1_B, PTR_W1_W, PTR_W2_B, PTR_W2_W,
    ROW_BIAS_W, ROW_GATE_B, ROW_GATE_W, ROW_LNM_B, ROW_LNM_G, ROW_LNP_B, ROW_LNP_G,
    ROW_OUT_B, ROW_OUT_W, ROW_QKV_W,
    TAE_BIAS_W, TAE_GATE_B, TAE_GATE_W, TAE_K_W, TAE_LN_B, TAE_LN_G, TAE_OUT_B, TAE_OUT_W, TAE_Q_W, TAE_V_W,
    TAS_BIAS_W, TAS_GATE_B, TAS_GATE_W, TAS_K_W, TAS_LN_B, TAS_LN_G, TAS_OUT_B, TAS_OUT_W, TAS_Q_W, TAS_V_W,
    TMI_G_B, TMI_G_W, TMI_GA_B, TMI_GA_W, TMI_GB_B, TMI_GB_W, TMI_LNI_B, TMI_LNI_G,
    TMI_LNO_B, TMI_LNO_G, TMI_OUT_B, TMI_OUT_W, TMI_PA_B, TMI_PA_W, TMI_PB_B, TMI_PB_W,
    TMO_G_B, TMO_G_W, TMO_GA_B, TMO_GA_W, TMO_GB_B, TMO_GB_W, TMO_LNI_B, TMO_LNI_G,
    TMO_LNO_B, TMO_LNO_G, TMO_OUT_B, TMO_OUT_W, TMO_PA_B, TMO_PA_W, TMO_PB_B, TMO_PB_W
};

extern "C" void kernel_launch(void* const* d_in, const int* in_sizes, int n_in,
                              void* d_out, int out_size, void* d_ws, size_t ws_size,
                              hipStream_t stream)
{
    (void)in_sizes; (void)n_in; (void)out_size; (void)ws_size;
    const int  S = 128, I = 256, E = 64, P = 128;
    const long SI = (long)S * I;       // 32768
    const long II = (long)I * I;       // 65536
    const float inv8   = 0.3535533906f;   // 1/sqrt(8)
    const float inv128 = 0.0883883476f;   // 1/sqrt(128)

    const float* msa_in  = (const float*)d_in[IN_MSA];
    const float* pair_in = (const float*)d_in[IN_PAIR];
    auto F = [&](int i) -> const float* { return (const float*)d_in[i]; };

    float* msa_o  = (float*)d_out;          // [S,I,E]
    float* pair_o = msa_o + SI * E;         // [I,I,P]

    char* wsb = (char*)d_ws;
    size_t off = 0;
    auto alloc = [&](size_t bytes) -> void* {
        void* p = wsb + off; off += (bytes + 255) & ~(size_t)255; return p; };
    auto wT16 = [&](int idx, long K, long N) -> _Float16* {
        _Float16* p = (_Float16*)alloc((size_t)(K * N) * sizeof(_Float16));
        cvtT16_kernel<<<gcount(K * N), 256, 0, stream>>>(F(idx), p, K, N);
        return p; };

    copy_kernel<<<gcount(SI * E), 256, 0, stream>>>(msa_in,  msa_o,  SI * E);
    copy_kernel<<<gcount(II * P), 256, 0, stream>>>(pair_in, pair_o, II * P);

    // ============================ row attention ============================
    off = 0;
    {
        _Float16* wbias = wT16(ROW_BIAS_W, 128, 8);
        _Float16* wqkv  = wT16(ROW_QKV_W,  64, 192);
        _Float16* wgate = wT16(ROW_GATE_W, 64, 64);
        _Float16* wout  = wT16(ROW_OUT_W,  64, 64);
        _Float16* m16   = (_Float16*)alloc(SI * E * 2);
        _Float16* z16   = (_Float16*)alloc(II * P * 2);
        float*    brow  = (float*)alloc(II * 8 * 4);
        _Float16* qkv16 = (_Float16*)alloc(SI * 192 * 2);
        float*    gate32= (float*)alloc(SI * 64 * 4);
        float*    o32   = (float*)alloc(SI * 64 * 4);
        float*    lg    = (float*)alloc((long)16 * 8 * 65536 * 4);
        _Float16* a16   = (_Float16*)alloc((long)16 * 8 * 65536 * 2);
        _Float16* gt16  = (_Float16*)alloc(SI * 64 * 2);

        ln_kernel<<<(unsigned)SI, 128, 0, stream>>>(msa_o,  F(ROW_LNM_G), F(ROW_LNM_B), m16, E);
        ln_kernel<<<(unsigned)II, 128, 0, stream>>>(pair_o, F(ROW_LNP_G), F(ROW_LNP_B), z16, P);
        gemm(stream, z16,128,1,0,0, wbias,1,128,0,0, brow,nullptr,8,1,0,0,
             nullptr,nullptr, II,8,128, 1,1, 1.f,0);
        gemm(stream, m16,64,1,0,0, wqkv,1,64,0,0, nullptr,qkv16,192,1,0,0,
             nullptr,nullptr, SI,192,64, 1,1, 1.f,0);
        gemm(stream, m16,64,1,0,0, wgate,1,64,0,0, gate32,nullptr,64,1,0,0,
             F(ROW_GATE_B),nullptr, SI,64,64, 1,1, 1.f,2);
        for (int s0 = 0; s0 < S; s0 += 16) {
            const long qb = (long)s0 * I * 192;
            gemm(stream, qkv16+qb,192,1,(long)I*192,8,
                         qkv16+qb+64,1,192,(long)I*192,8,
                 lg,nullptr,256,1,(long)8*65536,65536,
                 nullptr,nullptr, 256,256,8, 16*8,8, inv8,0);
            addbias_rowattn_kernel<<<gcount((long)16*8*65536), 256, 0, stream>>>(lg, brow, (long)16*8*65536);
            softmax_kernel<<<(unsigned)(16*8*256), 256, 0, stream>>>(lg, a16, 256, 256, 1);
            gemm(stream, a16,256,1,(long)8*65536,65536,
                         qkv16+qb+128,192,1,(long)I*192,8,
                 o32+(long)s0*I*64,nullptr,64,1,(long)I*64,8,
                 nullptr,nullptr, 256,8,256, 16*8,8, 1.f,0);
        }
        mul_ff16_kernel<<<gcount(SI*64), 256, 0, stream>>>(gate32, o32, gt16, SI * 64);
        gemm(stream, gt16,64,1,0,0, wout,1,64,0,0, msa_o,nullptr,64,1,0,0,
             F(ROW_OUT_B), msa_o, SI,64,64, 1,1, 1.f,0);
    }

    // ============================ global attention =========================
    off = 0;
    {
        _Float16* wqkv  = wT16(GLOB_QKV_W,  64, 80);
        _Float16* wgate = wT16(GLOB_GATE_W, 64, 64);
        _Float16* wout  = wT16(GLOB_OUT_W,  64, 64);
        _Float16* m16   = (_Float16*)alloc(SI * 64 * 2);
        float*    gq32  = (float*)alloc(SI * 80 * 4);
        _Float16* gq16  = (_Float16*)alloc(SI * 80 * 2);
        float*    ggate = (float*)alloc(SI * 64 * 4);
        _Float16* qm16  = (_Float16*)alloc((long)I * 64 * 2);
        float*    glog  = (float*)alloc((long)S * I * 8 * 4);
        _Float16* ga16  = (_Float16*)alloc((long)S * I * 8 * 2);
        float*    go32  = (float*)alloc((long)I * 64 * 4);
        _Float16* gt16  = (_Float16*)alloc(SI * 64 * 2);

        ln_kernel<<<(unsigned)SI, 128, 0, stream>>>(msa_o, F(GLOB_LN_G), F(GLOB_LN_B), m16, E);
        gemm(stream, m16,64,1,0,0, wqkv,1,64,0,0, gq32,gq16,80,1,0,0,
             nullptr,nullptr, SI,80,64, 1,1, 1.f,0);
        gemm(stream, m16,64,1,0,0, wgate,1,64,0,0, ggate,nullptr,64,1,0,0,
             F(GLOB_GATE_B),nullptr, SI,64,64, 1,1, 1.f,2);
        mean_s_kernel<<<gcount((long)I*64), 256, 0, stream>>>(gq32, qm16, S, I, 80, 64);
        gemm(stream, gq16+64,(long)I*80,1, 0,80,  qm16,8,1, 0,64,
             glog,nullptr,(long)I*8,1, 0,8, nullptr,nullptr, S,8,8, I,I, inv8,0);
        softmax_kernel<<<(unsigned)(I*8), 256, 0, stream>>>(glog, ga16, S, 1, (long)I*8);
        gemm(stream, gq16+72,1,(long)I*80, 0,80,  ga16,(long)I*8,1, 0,8,
             go32,nullptr,8,1, 0,64, nullptr,nullptr, 8,8,S, I,I, 1.f,0);
        bcast_mul16_kernel<<<gcount(SI*64), 256, 0, stream>>>(ggate, go32, gt16, SI*64, (long)I*64);
        gemm(stream, gt16,64,1,0,0, wout,1,64,0,0, msa_o,nullptr,64,1,0,0,
             F(GLOB_OUT_B), msa_o, SI,64,64, 1,1, 1.f,0);
    }

    // ============================ MSA transition ===========================
    off = 0;
    {
        _Float16* w1  = wT16(MTR_W1_W, 64, 256);
        _Float16* w2  = wT16(MTR_W2_W, 256, 64);
        _Float16* m16 = (_Float16*)alloc(SI * 64 * 2);
        _Float16* h16 = (_Float16*)alloc(SI * 256 * 2);
        ln_kernel<<<(unsigned)SI, 128, 0, stream>>>(msa_o, F(MTR_LN_G), F(MTR_LN_B), m16, E);
        gemm(stream, m16,64,1,0,0, w1,1,64,0,0, nullptr,h16,256,1,0,0,
             F(MTR_W1_B),nullptr, SI,256,64, 1,1, 1.f,1);
        gemm(stream, h16,256,1,0,0, w2,1,256,0,0, msa_o,nullptr,64,1,0,0,
             F(MTR_W2_B), msa_o, SI,64,256, 1,1, 1.f,0);
    }

    // ========================== outer product mean =========================
    off = 0;
    {
        _Float16* wa  = wT16(OPM_A_W, 64, 32);
        _Float16* wb  = wT16(OPM_B_W, 64, 32);
        _Float16* wo  = wT16(OPM_OUT_W, 1024, 128);
        _Float16* m16 = (_Float16*)alloc(SI * 64 * 2);
        _Float16* a16 = (_Float16*)alloc(SI * 32 * 2);
        _Float16* b16 = (_Float16*)alloc(SI * 32 * 2);
        float*    Oc  = (float*)alloc((long)32 * 256 * 1024 * 4);
        _Float16* Oc16= (_Float16*)alloc((long)32 * 256 * 1024 * 2);
        ln_kernel<<<(unsigned)SI, 128, 0, stream>>>(msa_o, F(OPM_LN_G), F(OPM_LN_B), m16, E);
        gemm(stream, m16,64,1,0,0, wa,1,64,0,0, nullptr,a16,32,1,0,0,
             F(OPM_A_B),nullptr, SI,32,64, 1,1, 1.f,0);
        gemm(stream, m16,64,1,0,0, wb,1,64,0,0, nullptr,b16,32,1,0,0,
             F(OPM_B_B),nullptr, SI,32,64, 1,1, 1.f,0);
        for (int i0 = 0; i0 < I; i0 += 32) {
            gemm(stream, a16+(long)i0*32, 1,(long)I*32, 32,0,
                         b16, (long)I*32,1, 0,32,
                 Oc,Oc16, 32,1, (long)256*1024,1024,
                 nullptr,nullptr, 32,32,S, 32*256,256, 1.f/S, 0);
            gemm(stream, Oc16,1024,1,0,0, wo,1,1024,0,0,
                 pair_o+(long)i0*256*128, nullptr,128,1,0,0,
                 F(OPM_OUT_B), pair_o+(long)i0*256*128, (long)32*256,128,1024, 1,1, 1.f,0);
        }
    }

    // ====================== triangle multiplications =======================
    struct TM { int ga_b, ga_w, pa_b, pa_w, gb_b, gb_w, pb_b, pb_w, g_b, g_w,
                    lni_b, lni_g, lno_b, lno_g, out_b, out_w, mode; };
    const TM tms[2] = {
        { TMO_GA_B,TMO_GA_W, TMO_PA_B,TMO_PA_W, TMO_GB_B,TMO_GB_W, TMO_PB_B,TMO_PB_W,
          TMO_G_B,TMO_G_W, TMO_LNI_B,TMO_LNI_G, TMO_LNO_B,TMO_LNO_G, TMO_OUT_B,TMO_OUT_W, 0 },
        { TMI_GA_B,TMI_GA_W, TMI_PA_B,TMI_PA_W, TMI_GB_B,TMI_GB_W, TMI_PB_B,TMI_PB_W,
          TMI_G_B,TMI_G_W, TMI_LNI_B,TMI_LNI_G, TMI_LNO_B,TMI_LNO_G, TMI_OUT_B,TMI_OUT_W, 1 } };
    for (int ti = 0; ti < 2; ++ti) {
        const TM& t = tms[ti];
        off = 0;
        _Float16* wga = wT16(t.ga_w, 128, 128);
        _Float16* wpa = wT16(t.pa_w, 128, 128);
        _Float16* wgb = wT16(t.gb_w, 128, 128);
        _Float16* wpb = wT16(t.pb_w, 128, 128);
        _Float16* wg  = wT16(t.g_w,  128, 128);
        _Float16* wo  = wT16(t.out_w,128, 128);
        _Float16* z16 = (_Float16*)alloc(II * 128 * 2);
        float*    t1  = (float*)alloc(II * 128 * 4);
        float*    t2  = (float*)alloc(II * 128 * 4);
        _Float16* a16 = (_Float16*)alloc(II * 128 * 2);
        _Float16* b16 = (_Float16*)alloc(II * 128 * 2);
        float*    g32 = (float*)alloc(II * 128 * 4);
        float*    o32 = (float*)alloc(II * 128 * 4);
        _Float16* ol16= (_Float16*)alloc(II * 128 * 2);
        float*    tmp = (float*)alloc(II * 128 * 4);

        ln_kernel<<<(unsigned)II, 128, 0, stream>>>(pair_o, F(t.lni_g), F(t.lni_b), z16, P);
        gemm(stream, z16,128,1,0,0, wga,1,128,0,0, t1,nullptr,128,1,0,0, F(t.ga_b),nullptr, II,128,128,1,1,1.f,2);
        gemm(stream, z16,128,1,0,0, wpa,1,128,0,0, t2,nullptr,128,1,0,0, F(t.pa_b),nullptr, II,128,128,1,1,1.f,0);
        mul_ff16_kernel<<<gcount(II*128), 256, 0, stream>>>(t1, t2, a16, II*128);
        gemm(stream, z16,128,1,0,0, wgb,1,128,0,0, t1,nullptr,128,1,0,0, F(t.gb_b),nullptr, II,128,128,1,1,1.f,2);
        gemm(stream, z16,128,1,0,0, wpb,1,128,0,0, t2,nullptr,128,1,0,0, F(t.pb_b),nullptr, II,128,128,1,1,1.f,0);
        mul_ff16_kernel<<<gcount(II*128), 256, 0, stream>>>(t1, t2, b16, II*128);
        gemm(stream, z16,128,1,0,0, wg,1,128,0,0, g32,nullptr,128,1,0,0, F(t.g_b),nullptr, II,128,128,1,1,1.f,2);
        if (t.mode == 0)
            gemm(stream, a16,(long)I*128,128, 0,1, b16,128,(long)I*128, 0,1,
                 o32,nullptr,(long)I*128,128, 0,1, nullptr,nullptr, I,I,I, 128,128, 1.f,0);
        else
            gemm(stream, a16,128,(long)I*128, 0,1, b16,(long)I*128,128, 0,1,
                 o32,nullptr,(long)I*128,128, 0,1, nullptr,nullptr, I,I,I, 128,128, 1.f,0);
        ln_kernel<<<(unsigned)II, 128, 0, stream>>>(o32, F(t.lno_g), F(t.lno_b), ol16, P);
        gemm(stream, ol16,128,1,0,0, wo,1,128,0,0, tmp,nullptr,128,1,0,0,
             F(t.out_b), nullptr, II,128,128, 1,1, 1.f,0);
        gated_add_kernel<<<gcount(II*128), 256, 0, stream>>>(pair_o, g32, tmp, II*128);
    }

    // ========================= triangle attentions =========================
    struct TA { int bias_w, gate_b, gate_w, k_w, ln_b, ln_g, out_b, out_w, q_w, v_w, start; };
    const TA tas2[2] = {
        { TAS_BIAS_W,TAS_GATE_B,TAS_GATE_W,TAS_K_W,TAS_LN_B,TAS_LN_G,TAS_OUT_B,TAS_OUT_W,TAS_Q_W,TAS_V_W, 1 },
        { TAE_BIAS_W,TAE_GATE_B,TAE_GATE_W,TAE_K_W,TAE_LN_B,TAE_LN_G,TAE_OUT_B,TAE_OUT_W,TAE_Q_W,TAE_V_W, 0 } };
    for (int ai = 0; ai < 2; ++ai) {
        const TA& t = tas2[ai];
        off = 0;
        _Float16* wq  = wT16(t.q_w,   128, 512);
        _Float16* wk  = wT16(t.k_w,   128, 512);
        _Float16* wv  = wT16(t.v_w,   128, 512);
        _Float16* wg  = wT16(t.gate_w,128, 512);
        _Float16* wbs = wT16(t.bias_w,128, 4);
        _Float16* wo  = wT16(t.out_w, 512, 128);
        _Float16* z16 = (_Float16*)alloc(II * 128 * 2);
        _Float16* q16 = (_Float16*)alloc(II * 512 * 2);
        _Float16* k16 = (_Float16*)alloc(II * 512 * 2);
        _Float16* v16 = (_Float16*)alloc(II * 512 * 2);
        _Float16* g16 = (_Float16*)alloc(II * 512 * 2);
        float*    bt  = (float*)alloc(II * 4 * 4);
        float*    lg  = (float*)alloc((long)32 * 4 * 65536 * 4);
        _Float16* at16= (_Float16*)alloc((long)32 * 4 * 65536 * 2);
        float*    och = (float*)alloc((long)32 * 256 * 512 * 4);
        _Float16* gt16= (_Float16*)alloc((long)32 * 256 * 512 * 2);
        float*    tmp = (float*)alloc((long)32 * 256 * 128 * 4);

        ln_kernel<<<(unsigned)II, 128, 0, stream>>>(pair_o, F(t.ln_g), F(t.ln_b), z16, P);
        gemm(stream, z16,128,1,0,0, wq,1,128,0,0, nullptr,q16,512,1,0,0, nullptr,nullptr, II,512,128,1,1,1.f,0);
        gemm(stream, z16,128,1,0,0, wk,1,128,0,0, nullptr,k16,512,1,0,0, nullptr,nullptr, II,512,128,1,1,1.f,0);
        gemm(stream, z16,128,1,0,0, wv,1,128,0,0, nullptr,v16,512,1,0,0, nullptr,nullptr, II,512,128,1,1,1.f,0);
        gemm(stream, z16,128,1,0,0, wg,1,128,0,0, nullptr,g16,512,1,0,0, F(t.gate_b),nullptr, II,512,128,1,1,1.f,2);
        gemm(stream, z16,128,1,0,0, wbs,1,128,0,0, bt,nullptr,4,1,0,0, nullptr,nullptr, II,4,128,1,1,1.f,0);

        for (int c0 = 0; c0 < I; c0 += 32) {
            if (t.start)
                gemm(stream, q16+(long)c0*256*512, 512,1, (long)256*512,128,
                             k16+(long)c0*256*512, 1,512, (long)256*512,128,
                     lg,nullptr, 256,1, (long)4*65536,65536,
                     nullptr,nullptr, 256,256,128, 32*4,4, inv128,0);
            else
                gemm(stream, q16+(long)c0*512, (long)256*512,1, 512,128,
                             k16+(long)c0*512, 1,(long)256*512, 512,128,
                     lg,nullptr, 256,1, (long)4*65536,65536,
                     nullptr,nullptr, 256,256,128, 32*4,4, inv128,0);
            addbias_tri_kernel<<<gcount((long)32*4*65536), 256, 0, stream>>>(
                lg, bt, t.start ? 0 : 1, (long)32*4*65536);
            softmax_kernel<<<(unsigned)(32*4*256), 256, 0, stream>>>(lg, at16, 256, 256, 1);
            if (t.start) {
                gemm(stream, at16,256,1,(long)4*65536,65536,
                             v16+(long)c0*256*512, 512,1, (long)256*512,128,
                     och,nullptr, 512,1, (long)256*512,128,
                     nullptr,nullptr, 256,128,256, 32*4,4, 1.f,0);
                mul_h16f_kernel<<<gcount((long)32*256*512), 256, 0, stream>>>(
                    g16+(long)c0*256*512, och, gt16, (long)32*256*512);
                gemm(stream, gt16,512,1,0,0, wo,1,512,0,0,
                     pair_o+(long)c0*256*128, nullptr,128,1,0,0,
                     F(t.out_b), pair_o+(long)c0*256*128, (long)32*256,128,512, 1,1, 1.f,0);
            } else {
                gemm(stream, at16,256,1,(long)4*65536,65536,
                             v16+(long)c0*512, (long)256*512,1, 512,128,
                     och,nullptr, (long)32*512,1, 512,128,
                     nullptr,nullptr, 256,128,256, 32*4,4, 1.f,0);
                mul_gate_jchunk_kernel<<<gcount((long)32*256*512), 256, 0, stream>>>(
                    g16, och, gt16, c0, 32, (long)32*256*512);
                gemm(stream, gt16,512,1,0,0, wo,1,512,0,0, tmp,nullptr,128,1,0,0,
                     F(t.out_b), nullptr, (long)32*256,128,512, 1,1, 1.f,0);
                scatter_add_jchunk_kernel<<<gcount((long)32*256*128), 256, 0, stream>>>(
                    pair_o, tmp, c0, 32, (long)32*256*128);
            }
        }
    }

    // ============================ pair transition ==========================
    off = 0;
    {
        _Float16* w1  = wT16(PTR_W1_W, 128, 512);
        _Float16* w2  = wT16(PTR_W2_W, 512, 128);
        _Float16* z16 = (_Float16*)alloc(II * 128 * 2);
        _Float16* h16 = (_Float16*)alloc(II * 512 * 2);
        ln_kernel<<<(unsigned)II, 128, 0, stream>>>(pair_o, F(PTR_LN_G), F(PTR_LN_B), z16, P);
        gemm(stream, z16,128,1,0,0, w1,1,128,0,0, nullptr,h16,512,1,0,0,
             F(PTR_W1_B),nullptr, II,512,128, 1,1, 1.f,1);
        gemm(stream, h16,512,1,0,0, w2,1,512,0,0, pair_o,nullptr,128,1,0,0,
             F(PTR_W2_B), pair_o, II,128,512, 1,1, 1.f,0);
    }
}
